// LinearSelfInteraction_27573690040902
// MI455X (gfx1250) — matrix operand order repaired and checked by reading, not compiled, not run on hardware
//
#include <hip/hip_runtime.h>

typedef __attribute__((ext_vector_type(16))) __bf16 v16bf;
typedef __attribute__((ext_vector_type(8)))  float  v8f;

#define DIM    2304
#define NNODES 50000

// Fragment = 8 contiguous VGPRs; fill with two 16B LDS loads.
union FragBF {
    uint4 q[2];
    v16bf v;
};

// Split-precision (bf16 hi/lo, 3 products) GEMM for one irrep block.
// Rows r = (n, c): element (r, k) lives at X[n*DIM + OFF + k*D + c].
template <int L>
__global__ __launch_bounds__(256)
void irrep_linear_kernel(const float* __restrict__ X,
                         const float* __restrict__ W,
                         float* __restrict__ Y)
{
    constexpr int D   = 2 * L + 1;
    constexpr int OFF = (L == 0) ? 0 : (L == 1) ? 256 : 1024;
    constexpr int ML  = NNODES * D;

    // K-pair dwords; rows padded to 20 dwords: 16B-aligned rows, stride 20 banks.
    __shared__ __align__(16) unsigned int sAhi[128][20];   // [row][kpair]
    __shared__ __align__(16) unsigned int sAlo[128][20];
    __shared__ __align__(16) unsigned int sBhi[128][20];   // [col][kpair] (transposed)
    __shared__ __align__(16) unsigned int sBlo[128][20];

    const int tid  = threadIdx.x;
    const int lane = tid & 31;
    const int wv   = tid >> 5;     // 8 waves
    const int half = lane >> 4;    // lane 0-15 vs 16-31
    const int l15  = lane & 15;
    const int wm   = wv >> 1;      // 0..3  -> 32-row stripe
    const int wn   = wv & 1;       // 0..1  -> 64-col stripe

    const int mb     = blockIdx.x >> 1;
    const int nb     = blockIdx.x & 1;
    const int m_base = mb * 128;
    const int n_base = nb * 128;

    // --- A staging map: thread = (kpair 0..15, row0 0..15), 8 row iterations
    const int a_kp = tid & 15;
    const int a_r0 = tid >> 4;
    unsigned int aBase[8];
    unsigned int aValid = 0u;
#pragma unroll
    for (int it = 0; it < 8; ++it) {
        int r = m_base + a_r0 + 16 * it;
        if (r < ML) {
            int n = r / D;
            int c = r - n * D;
            aBase[it] = (unsigned int)(n * DIM + OFF + c);
            aValid |= (1u << it);
        } else {
            aBase[it] = 0u;
        }
    }
    // --- B staging map: thread = (col 0..127, kpair0 0..1), 8 kpair iterations
    const int b_n   = tid & 127;
    const int b_kp0 = tid >> 7;

    v8f acc[2][4];
#pragma unroll
    for (int mt = 0; mt < 2; ++mt)
#pragma unroll
        for (int nt = 0; nt < 4; ++nt)
            acc[mt][nt] = (v8f){0.f, 0.f, 0.f, 0.f, 0.f, 0.f, 0.f, 0.f};

    for (int k0 = 0; k0 < 256; k0 += 32) {
        __syncthreads();   // previous iteration's fragment reads done

        // ---- stage A: fp32 -> bf16 hi/lo, packed K-pairs
        const int gk = k0 + 2 * a_kp;   // global K (mul index i), even
#pragma unroll
        for (int it = 0; it < 8; ++it) {
            float x0 = 0.f, x1 = 0.f;
            if ((aValid >> it) & 1u) {
                const float* p = X + aBase[it] + (unsigned int)gk * D;
                x0 = p[0];
                x1 = p[D];
            }
            __bf16 h0 = (__bf16)x0;
            __bf16 h1 = (__bf16)x1;
            __bf16 e0 = (__bf16)(x0 - (float)h0);
            __bf16 e1 = (__bf16)(x1 - (float)h1);
            union { __bf16 h[2]; unsigned int u; } ph, pl;
            ph.h[0] = h0; ph.h[1] = h1;
            pl.h[0] = e0; pl.h[1] = e1;
            const int row = a_r0 + 16 * it;
            sAhi[row][a_kp] = ph.u;
            sAlo[row][a_kp] = pl.u;
        }

        // ---- stage B: W[k][o] fp32 -> bf16 hi/lo, packed K-pairs, transposed in LDS
#pragma unroll
        for (int it = 0; it < 8; ++it) {
            const int kp  = b_kp0 + 2 * it;          // 0..15
            const int gk2 = k0 + 2 * kp;
            float w0 = W[gk2 * 256 + n_base + b_n];
            float w1 = W[(gk2 + 1) * 256 + n_base + b_n];
            __bf16 h0 = (__bf16)w0;
            __bf16 h1 = (__bf16)w1;
            __bf16 e0 = (__bf16)(w0 - (float)h0);
            __bf16 e1 = (__bf16)(w1 - (float)h1);
            union { __bf16 h[2]; unsigned int u; } ph, pl;
            ph.h[0] = h0; ph.h[1] = h1;
            pl.h[0] = e0; pl.h[1] = e1;
            sBhi[b_n][kp] = ph.u;
            sBlo[b_n][kp] = pl.u;
        }

        __syncthreads();

        // ---- load fragments with 16B LDS loads directly into operand VGPRs
        FragBF ahi[2], alo[2], bhi[4], blo[4];
#pragma unroll
        for (int mt = 0; mt < 2; ++mt) {
            const int m = wm * 32 + mt * 16 + l15;
            // A 16x32 layout: lanes 0-15 hold K-pairs {0..3, 8..11}; lanes 16-31 {4..7, 12..15}
            ahi[mt].q[0] = *reinterpret_cast<const uint4*>(&sAhi[m][4 * half]);
            ahi[mt].q[1] = *reinterpret_cast<const uint4*>(&sAhi[m][8 + 4 * half]);
            alo[mt].q[0] = *reinterpret_cast<const uint4*>(&sAlo[m][4 * half]);
            alo[mt].q[1] = *reinterpret_cast<const uint4*>(&sAlo[m][8 + 4 * half]);
        }
#pragma unroll
        for (int nt = 0; nt < 4; ++nt) {
            const int nn = wn * 64 + nt * 16 + l15;
            // B 32x16 layout: lanes 0-15 hold K-pairs {0..3, 8..11}; lanes 16-31 {4..7, 12..15}
            bhi[nt].q[0] = *reinterpret_cast<const uint4*>(&sBhi[nn][4 * half]);
            bhi[nt].q[1] = *reinterpret_cast<const uint4*>(&sBhi[nn][8 + 4 * half]);
            blo[nt].q[0] = *reinterpret_cast<const uint4*>(&sBlo[nn][4 * half]);
            blo[nt].q[1] = *reinterpret_cast<const uint4*>(&sBlo[nn][8 + 4 * half]);
        }

        // ---- 3-product split-precision accumulation (x_hi*w_hi + x_hi*w_lo + x_lo*w_hi)
#pragma unroll
        for (int mt = 0; mt < 2; ++mt) {
#pragma unroll
            for (int nt = 0; nt < 4; ++nt) {
                acc[mt][nt] = __builtin_amdgcn_wmma_f32_16x16x32_bf16(
                    false, ahi[mt].v, false, bhi[nt].v, (short)0, acc[mt][nt], false, false);
                acc[mt][nt] = __builtin_amdgcn_wmma_f32_16x16x32_bf16(
                    false, ahi[mt].v, false, blo[nt].v, (short)0, acc[mt][nt], false, false);
                acc[mt][nt] = __builtin_amdgcn_wmma_f32_16x16x32_bf16(
                    false, alo[mt].v, false, bhi[nt].v, (short)0, acc[mt][nt], false, false);
            }
        }
    }

    // ---- epilogue: scale by 1/sqrt(256) and scatter to interleaved layout
#pragma unroll
    for (int mt = 0; mt < 2; ++mt) {
#pragma unroll
        for (int v = 0; v < 8; ++v) {
            const int r = m_base + wm * 32 + mt * 16 + v + 8 * half;
            if (r < ML) {
                const int n = r / D;
                const int c = r - n * D;
                float* yrow = Y + n * DIM + OFF + c;
#pragma unroll
                for (int nt = 0; nt < 4; ++nt) {
                    const int o = n_base + wn * 64 + nt * 16 + l15;
                    yrow[o * D] = acc[mt][nt][v] * 0.0625f;
                }
            }
        }
    }
}

extern "C" void kernel_launch(void* const* d_in, const int* in_sizes, int n_in,
                              void* d_out, int out_size, void* d_ws, size_t ws_size,
                              hipStream_t stream) {
    (void)in_sizes; (void)n_in; (void)out_size; (void)d_ws; (void)ws_size;
    const float* X  = (const float*)d_in[0];
    // d_in[1] = Z_embeddings, unused by the reference
    const float* W0 = (const float*)d_in[2];
    const float* W1 = (const float*)d_in[3];
    const float* W2 = (const float*)d_in[4];
    float* Y = (float*)d_out;

    dim3 blk(256);
    const int mb0 = (NNODES * 1 + 127) / 128;   // 391
    const int mb1 = (NNODES * 3 + 127) / 128;   // 1172
    const int mb2 = (NNODES * 5 + 127) / 128;   // 1954
    irrep_linear_kernel<0><<<dim3(mb0 * 2), blk, 0, stream>>>(X, W0, Y);
    irrep_linear_kernel<1><<<dim3(mb1 * 2), blk, 0, stream>>>(X, W1, Y);
    irrep_linear_kernel<2><<<dim3(mb2 * 2), blk, 0, stream>>>(X, W2, Y);
}
